// FNOBlock_43173011259635
// MI455X (gfx1250) — matrix operand order repaired and checked by reading, not compiled, not run on hardware
//
#include <hip/hip_runtime.h>
#include <hip/hip_bf16.h>
#include <math.h>

#define N_GRID 65536
#define WIDTH 256
#define KMAX 64
#define NSLICES 64
#define SLICE_LEN (N_GRID / NSLICES)      // 1024 samples -> 32 k-steps per slice

typedef __attribute__((ext_vector_type(16))) _Float16 v16h;
typedef __attribute__((ext_vector_type(8)))  float    v8f;
typedef __attribute__((ext_vector_type(4)))  int      v4i_;

#define TWO_PI_OVER_N (6.28318530717958647692f / (float)N_GRID)

// ---------------------------------------------------------------------------
// CDNA5 async global->LDS copy (ASYNCcnt-tracked), with sync fallback.
// Probe result: builtin exists, arity 4, param0 = int4 addrspace(1)*.
// ---------------------------------------------------------------------------
#if defined(__has_builtin)
#if __has_builtin(__builtin_amdgcn_global_load_async_to_lds_b128) && \
    __has_builtin(__builtin_amdgcn_s_wait_asynccnt)
#define HAVE_ASYNC_LDS 1
#endif
#endif
#ifndef HAVE_ASYNC_LDS
#define HAVE_ASYNC_LDS 0
#endif

typedef __attribute__((address_space(1))) v4i_ gv4i;
typedef __attribute__((address_space(3))) v4i_ sv4i;

__device__ __forceinline__ void async_copy16(const _Float16* g, _Float16* l) {
#if HAVE_ASYNC_LDS
    __builtin_amdgcn_global_load_async_to_lds_b128(
        (gv4i*)(_Float16*)g, (sv4i*)l, 0, 0);
#else
    *(float4*)l = *(const float4*)g;
#endif
}

__device__ __forceinline__ void async_join() {
#if HAVE_ASYNC_LDS
    __builtin_amdgcn_s_wait_asynccnt(0);
#endif
    __syncthreads();
}

// ---------------------------------------------------------------------------
// Stage 0a: v -> f16 row-major (A operands) + f16 B-fragment order
//   B-frag layout: [nt=n/32][ct=c/16][lane][16],  lane=(c&15)+16*((n&31)>=16)
// ---------------------------------------------------------------------------
__global__ void cvt_v_kernel(const float* __restrict__ v,
                             _Float16* __restrict__ vh,
                             _Float16* __restrict__ vswz) {
    int i = blockIdx.x * blockDim.x + threadIdx.x;    // N_GRID*WIDTH threads
    int n = i >> 8, c = i & 255;
    _Float16 h = (_Float16)v[i];
    vh[i] = h;
    int nt = n >> 5, kk = n & 31;
    int ct = c >> 4, cn = c & 15;
    int lane = cn + (kk & 16);
    int idx  = kk & 15;
    vswz[((((size_t)nt * 16 + ct) * 32) + lane) * 16 + idx] = h;
}

// Stage 0b: W -> f16 B-fragment order: [ct][kstep=k/32][lane][16]
__global__ void cvt_w_kernel(const float* __restrict__ Wk,
                             _Float16* __restrict__ Wswz) {
    int i = blockIdx.x * blockDim.x + threadIdx.x;    // WIDTH*WIDTH threads
    int k = i >> 8, c = i & 255;
    int ct = c >> 4, cn = c & 15;
    int ks = k >> 5, kk = k & 31;
    int lane = cn + (kk & 16);
    int idx  = kk & 15;
    Wswz[(((size_t)(ct * 8 + ks) * 32) + lane) * 16 + idx] = (_Float16)Wk[i];
}

// ---------------------------------------------------------------------------
// Stage 0c: forward twiddles in A-fragment order: [kt][nt][lane][16]
//   element(lane,idx): m=lane&15, akb=8*(lane>=16), j = 2*akb+idx
//   k = kt*16+m, n = nt*32+j;  phi = (k*n mod N)*2pi/N  (exact in u32)
// ---------------------------------------------------------------------------
__global__ void tw_fwd_kernel(_Float16* __restrict__ tc, _Float16* __restrict__ ts) {
    unsigned i = blockIdx.x * blockDim.x + threadIdx.x;   // 1<<22 threads
    int idx = i & 15, lane = (i >> 4) & 31, nt = (i >> 9) & 2047, kt = (i >> 20) & 3;
    int m = lane & 15, akb = (lane >> 4) << 3;
    int j = 2 * akb + idx;
    unsigned k = (unsigned)(kt * 16 + m);
    unsigned n = (unsigned)(nt * 32 + j);
    float phi = (float)((k * n) & (N_GRID - 1)) * TWO_PI_OVER_N;
    float s, c;
    __sincosf(phi, &s, &c);
    tc[i] = (_Float16)c;
    ts[i] = (_Float16)(-s);
}

// Stage 0d: inverse twiddles in A-fragment order: [nt=n/16][ks=k/32][lane][16]
__global__ void tw_inv_kernel(_Float16* __restrict__ tc, _Float16* __restrict__ ts) {
    unsigned i = blockIdx.x * blockDim.x + threadIdx.x;   // 1<<22 threads
    int idx = i & 15, lane = (i >> 4) & 31, ks = (i >> 9) & 1, nt = (int)(i >> 10);
    int m = lane & 15, akb = (lane >> 4) << 3;
    int j = 2 * akb + idx;
    unsigned n = (unsigned)(nt * 16 + m);
    unsigned k = (unsigned)(ks * 32 + j);
    float phi = (float)((k * n) & (N_GRID - 1)) * TWO_PI_OVER_N;
    float s, c;
    __sincosf(phi, &s, &c);
    tc[i] = (_Float16)c;
    ts[i] = (_Float16)s;
}

// ---------------------------------------------------------------------------
// Stage 1: dense branch out = v @ W + b.
// Block: 8 waves, 128 rows x 64 cols. W slice (32KB, frag order) async-staged
// to LDS. Each wave: hoisted A frags (8 x 2 b128), 4 col-tiles x 8 WMMAs.
// ---------------------------------------------------------------------------
__global__ __launch_bounds__(256)
void dense_gemm_kernel(const _Float16* __restrict__ vh,
                       const _Float16* __restrict__ Wswz,
                       const float* __restrict__ bias,
                       float* __restrict__ out) {
    __shared__ __align__(32) _Float16 Bs[4 * 8 * 32 * 16];   // 32 KB
    int cb = blockIdx.x & 3;
    int rb = blockIdx.x >> 2;

    const _Float16* src = Wswz + (size_t)cb * (4 * 8 * 32 * 16);
    for (int t = threadIdx.x; t < (4 * 8 * 32 * 16) / 8; t += 256)
        async_copy16(src + t * 8, Bs + t * 8);
    async_join();

    int wave = threadIdx.x >> 5, lane = threadIdx.x & 31;
    int m0 = rb * 128 + wave * 16;
    int am = lane & 15, akb = (lane >> 4) << 3;
    const _Float16* Arow = vh + (size_t)(m0 + am) * WIDTH;

    v16h a[8];
#pragma unroll
    for (int ks = 0; ks < 8; ++ks) {
#pragma unroll
        for (int t = 0; t < 16; ++t) {
            int j = 2 * akb + t;
            a[ks][t] = Arow[ks * 32 + j];
        }
    }

    int bn = lane & 15, rowOff = (lane >> 4) * 8;
#pragma unroll
    for (int ctl = 0; ctl < 4; ++ctl) {
        v8f acc = {};
#pragma unroll
        for (int ks = 0; ks < 8; ++ks) {
            v16h b = *(const v16h*)&Bs[(((ctl * 8 + ks) * 32) + lane) * 16];
            acc = __builtin_amdgcn_wmma_f32_16x16x32_f16(false, a[ks], false, b,
                                                         (short)0, acc, false, false);
        }
        int c0 = (cb * 4 + ctl) * 16;
        float bb = bias[c0 + bn];
#pragma unroll
        for (int r = 0; r < 8; ++r)
            out[(size_t)(m0 + rowOff + r) * WIDTH + c0 + bn] = acc[r] + bb;
    }
}

// ---------------------------------------------------------------------------
// Stage 2: truncated forward DFT, slice partials (pure GEMM now).
// Block: 8 waves = 4 k-tiles x {cos,-sin}; v B-tiles (32KB) async-staged.
// ---------------------------------------------------------------------------
__global__ __launch_bounds__(256)
void dft_partial_kernel(const _Float16* __restrict__ vswz,
                        const _Float16* __restrict__ twc,
                        const _Float16* __restrict__ tws,
                        float* __restrict__ pre, float* __restrict__ pim) {
    __shared__ __align__(32) _Float16 Bs[32 * 32 * 16];      // 32 k-steps x 1KB
    int ct = blockIdx.x & 15;
    int slice = blockIdx.x >> 4;                             // 0..63

    for (int t = threadIdx.x; t < 2048; t += 256) {
        int ks = t >> 6;            // which 32x16 tile
        int s  = t & 63;            // 16-byte slot
        const _Float16* g = vswz +
            (((size_t)(slice * 32 + ks) * 16 + ct) * 32) * 16 + s * 8;
        async_copy16(g, Bs + ks * 512 + s * 8);
    }
    async_join();

    int wave = threadIdx.x >> 5, lane = threadIdx.x & 31;
    int kt = wave >> 1, comp = wave & 1;
    const _Float16* tw = comp ? tws : twc;

    v8f acc = {};
    for (int ks = 0; ks < 32; ++ks) {
        int nt = slice * 32 + ks;
        v16h a = *(const v16h*)&tw[(((size_t)kt * 2048 + nt) * 32 + lane) * 16];
        v16h b = *(const v16h*)&Bs[(ks * 32 + lane) * 16];
        acc = __builtin_amdgcn_wmma_f32_16x16x32_f16(false, a, false, b,
                                                     (short)0, acc, false, false);
    }

    int bn = lane & 15, rowOff = (lane >> 4) * 8;
    float* dst = comp ? pim : pre;
#pragma unroll
    for (int r = 0; r < 8; ++r) {
        int k = kt * 16 + rowOff + r;
        dst[((size_t)slice * KMAX + k) * WIDTH + ct * 16 + bn] = acc[r];
    }
}

// Deterministic second-stage reduction over slices.
__global__ void dft_reduce_kernel(const float* __restrict__ pre,
                                  const float* __restrict__ pim,
                                  float* __restrict__ vr, float* __restrict__ vi) {
    int i = blockIdx.x * blockDim.x + threadIdx.x;
    if (i >= KMAX * WIDTH) return;
    float sr = 0.f, si = 0.f;
    for (int s = 0; s < NSLICES; ++s) {
        sr += pre[(size_t)s * KMAX * WIDTH + i];
        si += pim[(size_t)s * KMAX * WIDTH + i];
    }
    vr[i] = sr;
    vi[i] = si;
}

// ---------------------------------------------------------------------------
// Stage 3: per-mode complex mixing + irfft weights, output in B-frag order:
//   layout [ct][ks=k/32][lane][16];  Br=w_k*Re, Bi=-w_k*Im
// ---------------------------------------------------------------------------
__global__ void mode_mix_kernel(const float* __restrict__ Rr,
                                const float* __restrict__ Ri,
                                const float* __restrict__ vr,
                                const float* __restrict__ vi,
                                _Float16* __restrict__ Br,
                                _Float16* __restrict__ Bi) {
    int i = blockIdx.x * blockDim.x + threadIdx.x;
    if (i >= KMAX * WIDTH) return;
    int k = i / WIDTH, o = i % WIDTH;
    const float* rr = Rr + ((size_t)k * WIDTH + o) * WIDTH;
    const float* ri = Ri + ((size_t)k * WIDTH + o) * WIDTH;
    const float* xr = vr + (size_t)k * WIDTH;
    const float* xi = vi + (size_t)k * WIDTH;
    float re = 0.f, im = 0.f;
    for (int t = 0; t < WIDTH; ++t) {
        re = fmaf(rr[t], xr[t], re) - ri[t] * xi[t];
        im = fmaf(rr[t], xi[t], im) + ri[t] * xr[t];
    }
    float w = (k == 0 ? 1.0f : 2.0f) / (float)N_GRID;
    int ct = o >> 4, cn = o & 15;
    int ks = k >> 5, kk = k & 31;
    int lane = cn + (kk & 16);
    int idx  = kk & 15;
    size_t pos = (((size_t)ct * 2 + ks) * 32 + lane) * 16 + idx;
    Br[pos] = (_Float16)(re * w);
    Bi[pos] = (_Float16)(-im * w);
}

// ---------------------------------------------------------------------------
// Stage 4: inverse truncated DFT fused with dense add + ReLU.
// Block: 8 waves, 128 rows x 64 cols; Br/Bi slices (16KB) async-staged.
// ---------------------------------------------------------------------------
__global__ __launch_bounds__(256)
void idft_combine_kernel(const _Float16* __restrict__ Brz,
                         const _Float16* __restrict__ Biz,
                         const _Float16* __restrict__ twc,
                         const _Float16* __restrict__ tws,
                         float* __restrict__ out) {
    __shared__ __align__(32) _Float16 Bs[2 * 4096];          // 16 KB (Br | Bi)
    int cb = blockIdx.x & 3;
    int rb = blockIdx.x >> 2;

    for (int t = threadIdx.x; t < 1024; t += 256) {
        int mat = t >> 9;
        int s   = t & 511;
        const _Float16* base = mat ? Biz : Brz;
        async_copy16(base + (size_t)cb * 4096 + s * 8, Bs + mat * 4096 + s * 8);
    }
    async_join();

    int wave = threadIdx.x >> 5, lane = threadIdx.x & 31;
    int n0 = rb * 128 + wave * 16;
    int nt = rb * 8 + wave;

    v16h acF[2], asF[2];
#pragma unroll
    for (int ks = 0; ks < 2; ++ks) {
        size_t off = (((size_t)nt * 2 + ks) * 32 + lane) * 16;
        acF[ks] = *(const v16h*)&twc[off];
        asF[ks] = *(const v16h*)&tws[off];
    }

    int bn = lane & 15, rowOff = (lane >> 4) * 8;
#pragma unroll
    for (int ctl = 0; ctl < 4; ++ctl) {
        v8f acc = {};
#pragma unroll
        for (int ks = 0; ks < 2; ++ks) {
            v16h br = *(const v16h*)&Bs[((ctl * 2 + ks) * 32 + lane) * 16];
            v16h bi = *(const v16h*)&Bs[4096 + ((ctl * 2 + ks) * 32 + lane) * 16];
            acc = __builtin_amdgcn_wmma_f32_16x16x32_f16(false, acF[ks], false, br,
                                                         (short)0, acc, false, false);
            acc = __builtin_amdgcn_wmma_f32_16x16x32_f16(false, asF[ks], false, bi,
                                                         (short)0, acc, false, false);
        }
        int c0 = (cb * 4 + ctl) * 16;
#pragma unroll
        for (int r = 0; r < 8; ++r) {
            size_t o = (size_t)(n0 + rowOff + r) * WIDTH + c0 + bn;
            float val = out[o] + acc[r];
            out[o] = val > 0.f ? val : 0.f;
        }
    }
}

// ---------------------------------------------------------------------------
extern "C" void kernel_launch(void* const* d_in, const int* in_sizes, int n_in,
                              void* d_out, int out_size, void* d_ws, size_t ws_size,
                              hipStream_t stream) {
    const float* v  = (const float*)d_in[0];
    const float* Wk = (const float*)d_in[1];
    const float* Wb = (const float*)d_in[2];
    const float* Rr = (const float*)d_in[3];
    const float* Ri = (const float*)d_in[4];
    float* out = (float*)d_out;

    char* ws = (char*)d_ws;
    size_t off = 0;
    _Float16* vh    = (_Float16*)(ws + off); off += (size_t)N_GRID * WIDTH * 2;      // 33.5 MB
    _Float16* vswz  = (_Float16*)(ws + off); off += (size_t)N_GRID * WIDTH * 2;      // 33.5 MB
    _Float16* Wswz  = (_Float16*)(ws + off); off += (size_t)WIDTH * WIDTH * 2;
    _Float16* twcF  = (_Float16*)(ws + off); off += (size_t)(1 << 22) * 2;           // 8 MB
    _Float16* twsF  = (_Float16*)(ws + off); off += (size_t)(1 << 22) * 2;
    _Float16* twcI  = (_Float16*)(ws + off); off += (size_t)(1 << 22) * 2;
    _Float16* twsI  = (_Float16*)(ws + off); off += (size_t)(1 << 22) * 2;
    float* pre      = (float*)(ws + off);    off += (size_t)NSLICES * KMAX * WIDTH * 4;
    float* pim      = (float*)(ws + off);    off += (size_t)NSLICES * KMAX * WIDTH * 4;
    float* vr       = (float*)(ws + off);    off += (size_t)KMAX * WIDTH * 4;
    float* vi       = (float*)(ws + off);    off += (size_t)KMAX * WIDTH * 4;
    _Float16* Brz   = (_Float16*)(ws + off); off += (size_t)KMAX * WIDTH * 2;
    _Float16* Biz   = (_Float16*)(ws + off); off += (size_t)KMAX * WIDTH * 2;

    // 0) repacks + twiddle tables
    cvt_v_kernel<<<(N_GRID * WIDTH) / 256, 256, 0, stream>>>(v, vh, vswz);
    cvt_w_kernel<<<(WIDTH * WIDTH) / 256, 256, 0, stream>>>(Wk, Wswz);
    tw_fwd_kernel<<<(1 << 22) / 256, 256, 0, stream>>>(twcF, twsF);
    tw_inv_kernel<<<(1 << 22) / 256, 256, 0, stream>>>(twcI, twsI);

    // 1) dense branch (WMMA)
    dense_gemm_kernel<<<(N_GRID / 128) * (WIDTH / 64), 256, 0, stream>>>(vh, Wswz, Wb, out);

    // 2) truncated forward DFT (WMMA, slice partials) + reduce
    dft_partial_kernel<<<16 * NSLICES, 256, 0, stream>>>(vswz, twcF, twsF, pre, pim);
    dft_reduce_kernel<<<(KMAX * WIDTH + 255) / 256, 256, 0, stream>>>(pre, pim, vr, vi);

    // 3) per-mode complex mixing -> fragment-order f16 B operands
    mode_mix_kernel<<<(KMAX * WIDTH + 255) / 256, 256, 0, stream>>>(Rr, Ri, vr, vi, Brz, Biz);

    // 4) inverse truncated DFT fused with add + ReLU (WMMA)
    idft_combine_kernel<<<(N_GRID / 128) * (WIDTH / 64), 256, 0, stream>>>(Brz, Biz, twcI, twsI, out);
}
